// SoftAttention_1881195675895
// MI455X (gfx1250) — matrix operand order repaired and checked by reading, not compiled, not run on hardware
//
#include <hip/hip_runtime.h>
#include <hip/hip_bf16.h>
#include <cstdint>

// ---------------------------------------------------------------------------
// Types for CDNA5 WMMA (wave32): v_wmma_f32_16x16x32_bf16
// ---------------------------------------------------------------------------
typedef __bf16 bf16_t;
typedef __attribute__((ext_vector_type(16))) __bf16 v16bf;
typedef __attribute__((ext_vector_type(8)))  __bf16 v8bf;
typedef __attribute__((ext_vector_type(8)))  float  v8f;

__device__ __forceinline__ v8f wmma_bf16(v16bf a, v16bf b, v8f c) {
    // 8 args: (neg_a, A, neg_b, B, c_mod, C, reuse_a, reuse_b)
    return __builtin_amdgcn_wmma_f32_16x16x32_bf16(false, a, false, b,
                                                   (short)0, c, false, false);
}

__device__ __forceinline__ v16bf combine16(v8bf lo, v8bf hi) {
    v16bf f;
#pragma unroll
    for (int i = 0; i < 8; ++i) { f[i] = lo[i]; f[i + 8] = hi[i]; }
    return f;
}

// ---------------------------------------------------------------------------
// CDNA5 async global->LDS copy (16B per lane), tracked with ASYNCcnt.
// Generic LDS pointer low 32 bits == LDS byte offset (LDS aperture mapping).
// ---------------------------------------------------------------------------
__device__ __forceinline__ void async_load_b128(uint32_t lds_byte_addr,
                                                const void* gaddr) {
    asm volatile("global_load_async_to_lds_b128 %0, %1, off"
                 :: "v"(lds_byte_addr),
                    "v"((unsigned long long)(uintptr_t)gaddr)
                 : "memory");
}
__device__ __forceinline__ void wait_async_le12() {
    asm volatile("s_wait_asynccnt 0xc" ::: "memory");
}
__device__ __forceinline__ void wait_async_0() {
    asm volatile("s_wait_asynccnt 0x0" ::: "memory");
}

// ---------------------------------------------------------------------------
// Cooperative batched bf16 GEMM with double-buffered async LDS staging.
//   C[b] (rowsM x colsN, f32) = A[b] (rowsM x K) * Bsrc[b]^T
//   Bsrc stored as (colsN x K) row-major (columns of mathematical B contiguous
//   in K).  Block = 128 threads = 4 waves arranged 2x2 over a 64x128 tile;
//   each wave computes a 32x64 sub-tile (2 A-frags x 4 B-frags, 8 f32 accums).
//   Per 64-K slice the block DMAs A(64x64) + B(128x64) into LDS with
//   global_load_async_to_lds_b128 (12 per thread), double-buffered so the DMA
//   of slice i+1 overlaps the 16 WMMAs on slice i.  LDS rows padded to 72 bf16
//   (144B) so ds_load_b128 fragment reads are bank-conflict-free.
// ---------------------------------------------------------------------------
#define LDS_PITCH 72
#define KSLICE    64

__global__ void __launch_bounds__(128)
wmma_gemm_bf16(const bf16_t* __restrict__ A, const bf16_t* __restrict__ Bsrc,
               float* __restrict__ C, int rowsM, int colsN, int K) {
    __shared__ bf16_t sA[2][64 * LDS_PITCH];
    __shared__ bf16_t sB[2][128 * LDS_PITCH];

    const int b      = blockIdx.y;
    const int tilesC = colsN >> 7;                 // 128-wide block tiles
    const int tile_r = blockIdx.x / tilesC;
    const int tile_c = blockIdx.x - tile_r * tilesC;
    const int rowB   = tile_r * 64;
    const int colB   = tile_c * 128;

    const int tid  = threadIdx.x;
    const int lane = tid & 31;
    const int w    = tid >> 5;                     // wave 0..3
    const int wr   = w >> 1, wc = w & 1;           // 2x2 wave grid
    const int r    = lane & 15, h = lane >> 4;

    const bf16_t* Ab = A    + (size_t)b * rowsM * K;
    const bf16_t* Bb = Bsrc + (size_t)b * colsN * K;
    float*        Cb = C    + (size_t)b * rowsM * colsN;

    // --- async fill of one K-slice (A: 64x64, B: 128x64) -------------------
    auto fill = [&](int buf, int kb) {
#pragma unroll
        for (int it = 0; it < 4; ++it) {           // A: 512 16B chunks
            int t = tid + it * 128;
            int row = t >> 3, q = t & 7;
            const void* g = Ab + (size_t)(rowB + row) * K + kb + q * 8;
            async_load_b128((uint32_t)(uintptr_t)&sA[buf][row * LDS_PITCH + q * 8], g);
        }
#pragma unroll
        for (int it = 0; it < 8; ++it) {           // B: 1024 16B chunks
            int t = tid + it * 128;
            int col = t >> 3, q = t & 7;
            const void* g = Bb + (size_t)(colB + col) * K + kb + q * 8;
            async_load_b128((uint32_t)(uintptr_t)&sB[buf][col * LDS_PITCH + q * 8], g);
        }
    };

    // --- fragment loads from LDS (documented CDNA5 WMMA layouts) -----------
    auto frag_a = [&](int buf, int rowloc, int ks) {  // A 16x32: K h*8 / 16+h*8
        const bf16_t* p = &sA[buf][rowloc * LDS_PITCH + ks];
        v8bf lo = *(const v8bf*)(p + h * 8);
        v8bf hi = *(const v8bf*)(p + 16 + h * 8);
        return combine16(lo, hi);
    };
    auto frag_b = [&](int buf, int colloc, int ks) {  // B 32x16: K h*8 / 16+h*8
        const bf16_t* p = &sB[buf][colloc * LDS_PITCH + ks];
        v8bf lo = *(const v8bf*)(p + h * 8);
        v8bf hi = *(const v8bf*)(p + 16 + h * 8);
        return combine16(lo, hi);
    };

    v8f acc[2][4] = {};

    auto compute = [&](int buf) {                  // 2 sub-steps x 8 WMMAs
#pragma unroll
        for (int s = 0; s < 2; ++s) {
            const int ks = s * 32;
            v16bf a0 = frag_a(buf, wr * 32 + r,       ks);
            v16bf a1 = frag_a(buf, wr * 32 + 16 + r,  ks);
            v16bf b0 = frag_b(buf, wc * 64 + r,       ks);
            v16bf b1 = frag_b(buf, wc * 64 + 16 + r,  ks);
            v16bf b2 = frag_b(buf, wc * 64 + 32 + r,  ks);
            v16bf b3 = frag_b(buf, wc * 64 + 48 + r,  ks);

            acc[0][0] = wmma_bf16(a0, b0, acc[0][0]);
            acc[0][1] = wmma_bf16(a0, b1, acc[0][1]);
            acc[0][2] = wmma_bf16(a0, b2, acc[0][2]);
            acc[0][3] = wmma_bf16(a0, b3, acc[0][3]);
            acc[1][0] = wmma_bf16(a1, b0, acc[1][0]);
            acc[1][1] = wmma_bf16(a1, b1, acc[1][1]);
            acc[1][2] = wmma_bf16(a1, b2, acc[1][2]);
            acc[1][3] = wmma_bf16(a1, b3, acc[1][3]);
        }
    };

    const int steps = K / KSLICE;

    fill(0, 0);
    for (int i = 0; i + 1 < steps; ++i) {          // branch-free steady state
        fill((i + 1) & 1, (i + 1) * KSLICE);       // prefetch next slice
        wait_async_le12();                         // retire current slice's 12
        __syncthreads();                           // slice visible block-wide
        compute(i & 1);
        __syncthreads();                           // reads done before refill
    }
    wait_async_0();                                // peeled last slice
    __syncthreads();
    compute((steps - 1) & 1);

    // C/D layout: VGPR e, lanes 0-15 -> M=e, N=lane; lanes 16-31 -> M=8+e
    const int row0 = rowB + wr * 32;
    const int col0 = colB + wc * 64;
#pragma unroll
    for (int i2 = 0; i2 < 2; ++i2) {
#pragma unroll
        for (int j = 0; j < 4; ++j) {
#pragma unroll
            for (int e = 0; e < 8; ++e) {
                int row = row0 + i2 * 16 + e + 8 * h;
                int col = col0 + 16 * j + r;
                Cb[(size_t)row * colsN + col] = acc[i2][j][e];
            }
        }
    }
}

// ---------------------------------------------------------------------------
// Convert fp32 -> bf16 and also produce a transposed bf16 copy (LDS tiled).
// block (32,8); each block handles a 32(rows) x 32(cols) tile.
// ---------------------------------------------------------------------------
__global__ void __launch_bounds__(256)
convert_transpose(const float* __restrict__ src, bf16_t* __restrict__ dst,
                  bf16_t* __restrict__ dstT, int R, int D) {
    __shared__ bf16_t tile[32][33];
    const int b  = blockIdx.z;
    const int d0 = blockIdx.x * 32;
    const int r0 = blockIdx.y * 32;
    const int tx = threadIdx.x, ty = threadIdx.y;
    const float* sb = src + (size_t)b * R * D;
#pragma unroll
    for (int i = 0; i < 4; ++i) {
        int rl = ty * 4 + i;
        float v = sb[(size_t)(r0 + rl) * D + d0 + tx];
        bf16_t bv = (bf16_t)v;
        dst[((size_t)b * R + r0 + rl) * D + d0 + tx] = bv;
        tile[tx][rl] = bv;                         // tile[d_local][r_local]
    }
    __syncthreads();
#pragma unroll
    for (int i = 0; i < 4; ++i) {
        int dl = ty * 4 + i;
        dstT[((size_t)b * D + d0 + dl) * R + r0 + tx] = tile[dl][tx];
    }
}

// ---------------------------------------------------------------------------
// Row softmax stats: one block per (b, n) row of scores; max + sum(exp).
// ---------------------------------------------------------------------------
__global__ void __launch_bounds__(256)
row_stats(const float* __restrict__ sc, float* __restrict__ rmax,
          float* __restrict__ rsum, int M) {
    __shared__ float red[256];
    const int bn = blockIdx.x;
    const float* row = sc + (size_t)bn * M;
    const int tid = threadIdx.x;

    float mx = -3.402823466e38f;
    for (int m = tid; m < M; m += 256) mx = fmaxf(mx, row[m]);
    red[tid] = mx; __syncthreads();
    for (int s = 128; s > 0; s >>= 1) {
        if (tid < s) red[tid] = fmaxf(red[tid], red[tid + s]);
        __syncthreads();
    }
    mx = red[0]; __syncthreads();

    float sm = 0.0f;
    for (int m = tid; m < M; m += 256) sm += expf(row[m] - mx);
    red[tid] = sm; __syncthreads();
    for (int s = 128; s > 0; s >>= 1) {
        if (tid < s) red[tid] += red[tid + s];
        __syncthreads();
    }
    if (tid == 0) { rmax[bn] = mx; rsum[bn] = red[0]; }
}

// ---------------------------------------------------------------------------
// Column softmax stats: thread per column m; coalesced row sweeps (L2 resident).
// ---------------------------------------------------------------------------
__global__ void __launch_bounds__(256)
col_stats(const float* __restrict__ sc, float* __restrict__ cmax,
          float* __restrict__ csum, int N, int M) {
    const int b = blockIdx.y;
    const int m = blockIdx.x * 256 + threadIdx.x;
    const float* base = sc + (size_t)b * N * M;

    float mx = -3.402823466e38f;
    for (int n = 0; n < N; ++n) mx = fmaxf(mx, base[(size_t)n * M + m]);
    float sm = 0.0f;
    for (int n = 0; n < N; ++n) sm += expf(base[(size_t)n * M + m] - mx);
    cmax[(size_t)b * M + m] = mx;
    csum[(size_t)b * M + m] = sm;
}

// ---------------------------------------------------------------------------
// Normalize: P_row[b,n,m] = exp(s - rmax[n]) / rsum[n]      (bf16, row-major)
//            P_colT[b,m,n] = exp(s - cmax[m]) / csum[m]     (bf16, transposed,
//                                                            LDS-tiled store)
// block (32,8); 32x32 score tile per block.
// ---------------------------------------------------------------------------
__global__ void __launch_bounds__(256)
normalize_probs(const float* __restrict__ sc, const float* __restrict__ rmax,
                const float* __restrict__ rsum, const float* __restrict__ cmax,
                const float* __restrict__ csum, bf16_t* __restrict__ Prow,
                bf16_t* __restrict__ PcolT, int N, int M) {
    __shared__ bf16_t tile[32][33];
    const int b  = blockIdx.z;
    const int m0 = blockIdx.x * 32;
    const int n0 = blockIdx.y * 32;
    const int tx = threadIdx.x, ty = threadIdx.y;
    const float* scb = sc + (size_t)b * N * M;
    const int m = m0 + tx;
    const float cm = cmax[(size_t)b * M + m];
    const float ci = 1.0f / csum[(size_t)b * M + m];
#pragma unroll
    for (int i = 0; i < 4; ++i) {
        int nl = ty * 4 + i;
        int n  = n0 + nl;
        float s  = scb[(size_t)n * M + m];
        float er = expf(s - rmax[(size_t)b * N + n]) / rsum[(size_t)b * N + n];
        Prow[((size_t)b * N + n) * M + m] = (bf16_t)er;
        float ec = expf(s - cm) * ci;
        tile[tx][nl] = (bf16_t)ec;                 // tile[m_local][n_local]
    }
    __syncthreads();
#pragma unroll
    for (int i = 0; i < 4; ++i) {
        int ml = ty * 4 + i;
        PcolT[((size_t)b * M + m0 + ml) * N + n0 + tx] = tile[ml][tx];
    }
}

// ---------------------------------------------------------------------------
// Host-side orchestration.
//   ws layout (bytes):
//     [0      , 32Mi ) x_bf16  [B][N][D]      (later overlapped by Prow)
//     [32Mi   , 64Mi ) y_bf16  [B][M][D]
//     [64Mi   , 96Mi ) xT_bf16 [B][D][N]
//     [96Mi   , 128Mi) yT_bf16 [B][D][M]
//     [128Mi  , 256Mi) scores  f32 [B][N][M]   (L2-resident, 128 MiB)
//     [0      , 64Mi ) Prow    bf16 [B][N][M]  (reuses x_bf16/y_bf16 region)
//     [256Mi  , 320Mi) PcolT   bf16 [B][M][N]
//     [320Mi  , +256K) rmax/rsum/cmax/csum
// ---------------------------------------------------------------------------
extern "C" void kernel_launch(void* const* d_in, const int* in_sizes, int n_in,
                              void* d_out, int out_size, void* d_ws, size_t ws_size,
                              hipStream_t stream) {
    (void)in_sizes; (void)n_in; (void)out_size; (void)ws_size;
    const int B = 8, N = 2048, M = 2048, D = 1024;

    const float* x = (const float*)d_in[0];
    const float* y = (const float*)d_in[1];
    float* out = (float*)d_out;

    char* ws = (char*)d_ws;
    const size_t MiB = 1ull << 20;
    bf16_t* xbf   = (bf16_t*)(ws);
    bf16_t* ybf   = (bf16_t*)(ws + 32 * MiB);
    bf16_t* xT    = (bf16_t*)(ws + 64 * MiB);
    bf16_t* yT    = (bf16_t*)(ws + 96 * MiB);
    float*  sc    = (float*) (ws + 128 * MiB);
    bf16_t* Prow  = (bf16_t*)(ws);               // reuse: x/y bf16 dead after scores
    bf16_t* PcolT = (bf16_t*)(ws + 256 * MiB);
    float*  rmax  = (float*) (ws + 320 * MiB);
    float*  rsum  = rmax + (size_t)B * N;
    float*  cmax  = rsum + (size_t)B * N;
    float*  csum  = cmax + (size_t)B * M;

    dim3 blkT(32, 8);
    convert_transpose<<<dim3(D / 32, N / 32, B), blkT, 0, stream>>>(x, xbf, xT, N, D);
    convert_transpose<<<dim3(D / 32, M / 32, B), blkT, 0, stream>>>(y, ybf, yT, M, D);

    // scores[b] = x_bf16[b] (NxD) @ y_bf16[b]^T  -> f32 [N x M]
    {
        int blocks = (N / 64) * (M / 128);
        wmma_gemm_bf16<<<dim3(blocks, B), 128, 0, stream>>>(xbf, ybf, sc, N, M, D);
    }

    row_stats<<<B * N, 256, 0, stream>>>(sc, rmax, rsum, M);
    col_stats<<<dim3(M / 256, B), 256, 0, stream>>>(sc, cmax, csum, N, M);
    normalize_probs<<<dim3(M / 32, N / 32, B), blkT, 0, stream>>>(
        sc, rmax, rsum, cmax, csum, Prow, PcolT, N, M);

    // A[b] = Prow[b] (NxM) @ y[b] (MxD)   : B-operand source = yT (D x M)
    {
        int blocks = (N / 64) * (D / 128);
        wmma_gemm_bf16<<<dim3(blocks, B), 128, 0, stream>>>(Prow, yT, out, N, D, M);
    }
    // Bout[b] = PcolT[b] (MxN) @ x[b] (NxD): B-operand source = xT (D x N)
    {
        int blocks = (M / 64) * (D / 128);
        wmma_gemm_bf16<<<dim3(blocks, B), 128, 0, stream>>>(
            PcolT, xT, out + (size_t)B * N * D, M, D, N);
    }
}
